// LSTM_61203283968689
// MI455X (gfx1250) — matrix operand order repaired and checked by reading, not compiled, not run on hardware
//
#include <hip/hip_runtime.h>
#include <math.h>

typedef _Float16 v16h __attribute__((ext_vector_type(16)));
typedef _Float16 v8h  __attribute__((ext_vector_type(8)));
typedef float    v8f  __attribute__((ext_vector_type(8)));

#define IN_DIM 22
#define HID    64
#define BATCH  1024
#define SEQ    512
#define OUTD   4
#define BTILE  16
#define HSTR   72   // padded LDS row stride (halves); 144B rows -> 16B aligned, bank-spread

// Branch-free activations on native TRANS ops (v_exp_f32 == exp2, v_rcp_f32).
// TRANS co-executes with WMMA on CDNA5, keeping the recurrence critical path tight.
__device__ __forceinline__ float sigf(float x) {
  return __builtin_amdgcn_rcpf(1.0f + __builtin_amdgcn_exp2f(x * -1.442695041f));
}
__device__ __forceinline__ float tanh_fast(float x) {
  // tanh(x) = 2*sigmoid(2x) - 1
  return 2.0f * __builtin_amdgcn_rcpf(1.0f + __builtin_amdgcn_exp2f(x * -2.885390082f)) - 1.0f;
}

// B-matrix fragment (32x16 f16) for V_WMMA_F32_16X16X32_F16, loaded from a
// row-major (256 x rowlen) f32 weight matrix (row = gate column => per-lane
// contiguous K run). kbase = kc*32 + laneHi*16. Pads K beyond rowlen with 0.
__device__ __forceinline__ v16h load_bfrag(const float* __restrict__ W, int rowlen,
                                           int n, int kbase) {
  const float* row = W + (size_t)n * rowlen;
  v16h f;
#pragma unroll
  for (int e = 0; e < 16; ++e) {
    int k = kbase + e;
    f[e] = (k < rowlen) ? (_Float16)row[k] : (_Float16)0.0f;
  }
  return f;
}

// A-matrix fragment (16x32 f16) chunk kc from a 16xHSTR row-major f16 LDS block.
// Layout: lanes<16 -> K = kc*32 + {0..15}; lanes>=16 -> +16.
__device__ __forceinline__ v16h load_afrag_lds(const _Float16* hrow,
                                               int laneN, int laneHi, int kc) {
  const _Float16* base = hrow + laneN * HSTR + kc * 32 + laneHi * 16;
  v8h lo = *(const v8h*)(base);
  v8h hi = *(const v8h*)(base + 8);
  v16h a;
#pragma unroll
  for (int e = 0; e < 8; ++e) { a[e] = lo[e]; a[8 + e] = hi[e]; }
  return a;
}

// Layer-0 A fragment straight from x (B, 22, T) f32, zero-padded K 22->32.
__device__ __forceinline__ v16h load_xfrag(const float* __restrict__ x, int b0,
                                           int laneN, int laneHi, int t) {
  const float* xb = x + ((size_t)(b0 + laneN) * IN_DIM) * SEQ + t;
  v16h a;
#pragma unroll
  for (int e = 0; e < 16; ++e) {
    int k = laneHi * 16 + e;
    a[e] = (k < IN_DIM) ? (_Float16)xb[(size_t)k * SEQ] : (_Float16)0.0f;
  }
  return a;
}

// Recurrent-layer gates: acc[s] = h_in @ Wih^T + h_prev @ Whh^T for the wave's
// four gate subtiles (i,f,g,o of its hidden block).
__device__ __forceinline__ void mm_layer(v8f (&acc)[4], const _Float16* ain,
                                         const _Float16* ahh,
                                         const v16h (&fih)[4][2], const v16h (&fhh)[4][2],
                                         int laneN, int laneHi) {
  v16h i0 = load_afrag_lds(ain, laneN, laneHi, 0);
  v16h i1 = load_afrag_lds(ain, laneN, laneHi, 1);
  v16h h0 = load_afrag_lds(ahh, laneN, laneHi, 0);
  v16h h1 = load_afrag_lds(ahh, laneN, laneHi, 1);
#pragma unroll
  for (int s = 0; s < 4; ++s) {
    v8f a = {};
    a = __builtin_amdgcn_wmma_f32_16x16x32_f16(false, i0, false, fih[s][0], (short)0, a, false, false);
    a = __builtin_amdgcn_wmma_f32_16x16x32_f16(false, i1, false, fih[s][1], (short)0, a, false, false);
    a = __builtin_amdgcn_wmma_f32_16x16x32_f16(false, h0, false, fhh[s][0], (short)0, a, false, false);
    a = __builtin_amdgcn_wmma_f32_16x16x32_f16(false, h1, false, fhh[s][1], (short)0, a, false, false);
    acc[s] = a;
  }
}

// LSTM cell update: all 4 gates of hidden unit j sit in the same lane/element.
__device__ __forceinline__ void ew_store(const v8f (&acc)[4], float (&c)[8],
                                         const float (&b)[4], _Float16* hout,
                                         int laneN, int laneHi, int j0) {
#pragma unroll
  for (int v = 0; v < 8; ++v) {
    float ig = sigf(acc[0][v] + b[0]);
    float fg = sigf(acc[1][v] + b[1]);
    float gg = tanh_fast(acc[2][v] + b[2]);
    float og = sigf(acc[3][v] + b[3]);
    float cc = fg * c[v] + ig * gg;
    c[v] = cc;
    hout[(laneHi * 8 + v) * HSTR + (j0 + laneN)] = (_Float16)(og * tanh_fast(cc));
  }
}

__global__ __launch_bounds__(128)
void lstm_fused(const float* __restrict__ x,
                const float* __restrict__ wih0, const float* __restrict__ whh0,
                const float* __restrict__ bih0, const float* __restrict__ bhh0,
                const float* __restrict__ wih1, const float* __restrict__ whh1,
                const float* __restrict__ bih1, const float* __restrict__ bhh1,
                const float* __restrict__ wih2, const float* __restrict__ whh2,
                const float* __restrict__ bih2, const float* __restrict__ bhh2,
                const float* __restrict__ wfc, const float* __restrict__ bfc,
                float* __restrict__ out) {
  __shared__ alignas(16) _Float16 hbuf[3][2][BTILE][HSTR];  // [layer][parity][row][j]

  const int tid    = threadIdx.x;
  const int lane   = tid & 31;
  const int wv     = tid >> 5;       // wave 0..3 -> hidden block
  const int laneN  = lane & 15;
  const int laneHi = lane >> 4;
  const int j0     = wv * 16;
  const int b0     = blockIdx.x * BTILE;

  for (int i = tid; i < 3 * 2 * BTILE * HSTR; i += blockDim.x)
    ((_Float16*)hbuf)[i] = (_Float16)0.0f;

  // Loop-invariant weight B-fragments: held in VGPRs for all 512 steps
  // (uses the CDNA5 extended VGPR file via s_set_vgpr_msb).
  v16h fih0[4], fhh0[4][2], fih1[4][2], fhh1[4][2], fih2[4][2], fhh2[4][2];
#pragma unroll
  for (int s = 0; s < 4; ++s) {
    const int n = s * HID + j0 + laneN;        // gate column owned by this lane
    fih0[s] = load_bfrag(wih0, IN_DIM, n, laneHi * 16);
#pragma unroll
    for (int kc = 0; kc < 2; ++kc) {
      const int kb = kc * 32 + laneHi * 16;
      fhh0[s][kc] = load_bfrag(whh0, HID, n, kb);
      fih1[s][kc] = load_bfrag(wih1, HID, n, kb);
      fhh1[s][kc] = load_bfrag(whh1, HID, n, kb);
      fih2[s][kc] = load_bfrag(wih2, HID, n, kb);
      fhh2[s][kc] = load_bfrag(whh2, HID, n, kb);
    }
  }

  // Per-lane combined biases (b_ih + b_hh) for the 4 gates of hidden unit j.
  float bias0[4], bias1[4], bias2[4];
  {
    const int j = j0 + laneN;
#pragma unroll
    for (int g = 0; g < 4; ++g) {
      bias0[g] = bih0[g * HID + j] + bhh0[g * HID + j];
      bias1[g] = bih1[g * HID + j] + bhh1[g * HID + j];
      bias2[g] = bih2[g * HID + j] + bhh2[g * HID + j];
    }
  }

  float c0[8], c1[8], c2[8];
#pragma unroll
  for (int v = 0; v < 8; ++v) { c0[v] = 0.0f; c1[v] = 0.0f; c2[v] = 0.0f; }

  __syncthreads();

  for (int t = 0; t < SEQ; ++t) {
    const int cur = t & 1, prv = cur ^ 1;
    if (t + 1 < SEQ)
      __builtin_prefetch(x + ((size_t)(b0 + laneN) * IN_DIM) * SEQ + (t + 1), 0, 0);

    // ---- layer 0: gates = x_t @ Wih0^T + h0 @ Whh0^T ----
    v8f acc[4];
    {
      v16h ax = load_xfrag(x, b0, laneN, laneHi, t);
      v16h a0 = load_afrag_lds(&hbuf[0][prv][0][0], laneN, laneHi, 0);
      v16h a1 = load_afrag_lds(&hbuf[0][prv][0][0], laneN, laneHi, 1);
#pragma unroll
      for (int s = 0; s < 4; ++s) {
        v8f a = {};
        a = __builtin_amdgcn_wmma_f32_16x16x32_f16(false, ax, false, fih0[s],    (short)0, a, false, false);
        a = __builtin_amdgcn_wmma_f32_16x16x32_f16(false, a0, false, fhh0[s][0], (short)0, a, false, false);
        a = __builtin_amdgcn_wmma_f32_16x16x32_f16(false, a1, false, fhh0[s][1], (short)0, a, false, false);
        acc[s] = a;
      }
    }
    ew_store(acc, c0, bias0, &hbuf[0][cur][0][0], laneN, laneHi, j0);
    __syncthreads();

    // ---- layer 1 ----
    mm_layer(acc, &hbuf[0][cur][0][0], &hbuf[1][prv][0][0], fih1, fhh1, laneN, laneHi);
    ew_store(acc, c1, bias1, &hbuf[1][cur][0][0], laneN, laneHi, j0);
    __syncthreads();

    // ---- layer 2 ----
    mm_layer(acc, &hbuf[1][cur][0][0], &hbuf[2][prv][0][0], fih2, fhh2, laneN, laneHi);
    ew_store(acc, c2, bias2, &hbuf[2][cur][0][0], laneN, laneHi, j0);
    __syncthreads();
  }

  // ---- final FC: out = h2[:, T-1, :] @ w_fc^T + b_fc ----
  const int lastpar = (SEQ - 1) & 1;
  if (tid < BTILE * OUTD) {
    const int r = tid >> 2, o = tid & 3;
    float s = bfc[o];
#pragma unroll
    for (int j = 0; j < HID; ++j)
      s += (float)hbuf[2][lastpar][r][j] * wfc[o * HID + j];
    out[(size_t)(b0 + r) * OUTD + o] = s;
  }
}

extern "C" void kernel_launch(void* const* d_in, const int* in_sizes, int n_in,
                              void* d_out, int out_size, void* d_ws, size_t ws_size,
                              hipStream_t stream) {
  (void)in_sizes; (void)n_in; (void)d_ws; (void)ws_size; (void)out_size;
  const float* p[15];
  for (int i = 0; i < 15; ++i) p[i] = (const float*)d_in[i];
  dim3 grid(BATCH / BTILE), block(128);
  hipLaunchKernelGGL(lstm_fused, grid, block, 0, stream,
                     p[0], p[1], p[2], p[3], p[4], p[5], p[6], p[7], p[8],
                     p[9], p[10], p[11], p[12], p[13], p[14], (float*)d_out);
}
